// GATEncoder_29291676959175
// MI455X (gfx1250) — compile-verified, matrix operand order checked by reading, not run on hardware
//
#include <hip/hip_runtime.h>
#include <hip/hip_bf16.h>

#define IN_DIM 1024
#define HID 64
#define HEADS 4
#define F1 (HEADS * HID)     // 256
#define OUT_DIM 128
#define NEG_SLOPE 0.2f

typedef float v8f __attribute__((ext_vector_type(8)));
typedef __bf16 v16bf __attribute__((ext_vector_type(16)));
typedef unsigned short u16x16 __attribute__((ext_vector_type(16)));

// ---------- helpers ----------
__device__ __forceinline__ unsigned short f2bf(float f) {
    unsigned u = __float_as_uint(f);
    unsigned r = u + 0x7FFFu + ((u >> 16) & 1u);   // round-to-nearest-even
    return (unsigned short)(r >> 16);
}
__device__ __forceinline__ unsigned enc_f32(float f) {   // order-preserving uint key
    unsigned u = __float_as_uint(f);
    return (u & 0x80000000u) ? ~u : (u | 0x80000000u);
}
__device__ __forceinline__ float dec_f32(unsigned k) {
    unsigned u = (k & 0x80000000u) ? (k ^ 0x80000000u) : ~k;
    return __uint_as_float(u);
}
__device__ __forceinline__ void edge_nodes(const long long* __restrict__ ei,
                                           long long E0, long long edge,
                                           int& src, int& dst) {
    if (edge < E0) { src = (int)ei[edge]; dst = (int)ei[E0 + edge]; }
    else           { src = dst = (int)(edge - E0); }
}

// ---------- fill ----------
__global__ void k_fill_f32(float* p, float v, size_t n) {
    size_t i = (size_t)blockIdx.x * blockDim.x + threadIdx.x;
    size_t s = (size_t)gridDim.x * blockDim.x;
    for (; i < n; i += s) p[i] = v;
}
__global__ void k_fill_u32(unsigned* p, unsigned v, size_t n) {
    size_t i = (size_t)blockIdx.x * blockDim.x + threadIdx.x;
    size_t s = (size_t)gridDim.x * blockDim.x;
    for (; i < n; i += s) p[i] = v;
}

// ---------- transpose+convert weights: W[K][Nc] f32 -> Wt[Nc][K] bf16 ----------
__global__ void k_transpose_bf16(const float* __restrict__ W, unsigned short* __restrict__ Wt,
                                 int K, int Nc) {
    int t = blockIdx.x * blockDim.x + threadIdx.x;
    if (t >= K * Nc) return;
    int nn = t / K, kk = t % K;
    Wt[t] = f2bf(W[(size_t)kk * Nc + nn]);
}

// ---------- bf16 WMMA GEMM: C[M][Nc] = A[M][K](f32) * Bt[Nc][K](bf16)^T ----------
__global__ __launch_bounds__(256)
void k_gemm_bf16_wmma(const float* __restrict__ A, const unsigned short* __restrict__ Bt,
                      float* __restrict__ C, int M, int K, int Nc) {
    __shared__ unsigned short As[128][48];   // 32 K + pad, row = 96B (32B aligned)
    __shared__ unsigned short Bs[128][48];
    const int tid  = threadIdx.x;
    const int lane = tid & 31;
    const int wave = tid >> 5;
    const int wm = wave >> 1;            // 0..3
    const int wn = wave & 1;             // 0..1
    const int m0 = blockIdx.x * 128;
    const int n0 = blockIdx.y * 128;

    v8f acc[2][4];
#pragma unroll
    for (int i = 0; i < 2; ++i)
#pragma unroll
        for (int j = 0; j < 4; ++j) acc[i][j] = (v8f)0.0f;

    for (int k0 = 0; k0 < K; k0 += 32) {
        // stage A tile 128x32 f32 -> bf16 (4 float4 chunks per thread)
#pragma unroll
        for (int it = 0; it < 4; ++it) {
            int c   = tid + it * 256;          // 0..1023
            int row = c >> 3;
            int col = (c & 7) << 2;
            float4 v = {0.f, 0.f, 0.f, 0.f};
            int gr = m0 + row;
            if (gr < M) v = *(const float4*)(A + (size_t)gr * K + k0 + col);
            As[row][col + 0] = f2bf(v.x);
            As[row][col + 1] = f2bf(v.y);
            As[row][col + 2] = f2bf(v.z);
            As[row][col + 3] = f2bf(v.w);
        }
        // stage B tile 128(N) x 32(K) bf16 (16B chunks)
#pragma unroll
        for (int it = 0; it < 2; ++it) {
            int c   = tid + it * 256;          // 0..511
            int row = c >> 2;
            int col = (c & 3) << 3;
            uint4 w = *(const uint4*)(Bt + (size_t)(n0 + row) * K + k0 + col);
            *(uint4*)&Bs[row][col] = w;
        }
        if (k0 + 32 < K) {
            int gr = m0 + (tid >> 1);
            if (gr < M) __builtin_prefetch(A + (size_t)gr * K + k0 + 32 + (tid & 1) * 16);
        }
        __syncthreads();

        const int ks = (lane >> 4) << 4;       // 0 or 16
        v16bf afr[2], bfr[4];
#pragma unroll
        for (int mi = 0; mi < 2; ++mi) {
            int r = wm * 32 + mi * 16 + (lane & 15);
            afr[mi] = __builtin_bit_cast(v16bf, *(const u16x16*)&As[r][ks]);
        }
#pragma unroll
        for (int ni = 0; ni < 4; ++ni) {
            int r = wn * 64 + ni * 16 + (lane & 15);
            bfr[ni] = __builtin_bit_cast(v16bf, *(const u16x16*)&Bs[r][ks]);
        }
#pragma unroll
        for (int mi = 0; mi < 2; ++mi)
#pragma unroll
            for (int ni = 0; ni < 4; ++ni)
                acc[mi][ni] = __builtin_amdgcn_wmma_f32_16x16x32_bf16(
                    false, afr[mi], false, bfr[ni], (short)0, acc[mi][ni], false, false);
        __syncthreads();
    }

    // epilogue: C/D layout — VGPR r, lanes 0-15: M=r, lanes 16-31: M=r+8; N = lane%16
#pragma unroll
    for (int mi = 0; mi < 2; ++mi) {
#pragma unroll
        for (int ni = 0; ni < 4; ++ni) {
            int colg = n0 + wn * 64 + ni * 16 + (lane & 15);
#pragma unroll
            for (int r = 0; r < 8; ++r) {
                int rowg = m0 + wm * 32 + mi * 16 + r + ((lane >> 4) << 3);
                if (rowg < M) C[(size_t)rowg * Nc + colg] = acc[mi][ni][r];
            }
        }
    }
}

// ---------- per-node attention halves: alpha_s/alpha_d [N,H] ----------
__global__ void k_alpha(const float* __restrict__ h, const float* __restrict__ asrc,
                        const float* __restrict__ adst, float* __restrict__ outs,
                        float* __restrict__ outd, int N, int H, int D) {
    int t = blockIdx.x * blockDim.x + threadIdx.x;
    if (t >= N * H) return;
    int n = t / H, hh = t % H;
    const float* hp = h + (size_t)n * H * D + (size_t)hh * D;
    const float* sa = asrc + hh * D;
    const float* da = adst + hh * D;
    float s1 = 0.f, s2 = 0.f;
    for (int i = 0; i < D; ++i) { float v = hp[i]; s1 += v * sa[i]; s2 += v * da[i]; }
    outs[t] = s1; outd[t] = s2;
}

// ---------- edge pass A: segment max via ordered-uint atomicMax ----------
__global__ void k_edge_max(const long long* __restrict__ ei, long long E0, long long E, int H,
                           const float* __restrict__ as, const float* __restrict__ ad,
                           unsigned* __restrict__ mkey) {
    long long t = (long long)blockIdx.x * blockDim.x + threadIdx.x;
    if (t >= E * H) return;
    long long edge = t / H; int hh = (int)(t % H);
    int src, dst; edge_nodes(ei, E0, edge, src, dst);
    float e = as[(size_t)src * H + hh] + ad[(size_t)dst * H + hh];
    e = e > 0.f ? e : NEG_SLOPE * e;
    atomicMax(&mkey[(size_t)dst * H + hh], enc_f32(e));
}

// ---------- edge pass B: w = exp(e - m), segment sum ----------
__global__ void k_edge_expsum(const long long* __restrict__ ei, long long E0, long long E, int H,
                              const float* __restrict__ as, const float* __restrict__ ad,
                              const unsigned* __restrict__ mkey,
                              float* __restrict__ wbuf, float* __restrict__ ssum) {
    long long t = (long long)blockIdx.x * blockDim.x + threadIdx.x;
    if (t >= E * H) return;
    long long edge = t / H; int hh = (int)(t % H);
    int src, dst; edge_nodes(ei, E0, edge, src, dst);
    float e = as[(size_t)src * H + hh] + ad[(size_t)dst * H + hh];
    e = e > 0.f ? e : NEG_SLOPE * e;
    float m = dec_f32(mkey[(size_t)dst * H + hh]);
    float w = __expf(e - m);
    wbuf[(size_t)edge * H + hh] = w;
    atomicAdd(&ssum[(size_t)dst * H + hh], w);
}

// ---------- edge pass C: out[dst] += h[src] * (w / s[dst]) ; wave per edge ----------
__global__ void k_edge_agg(const long long* __restrict__ ei, long long E0, long long E,
                           int H, int F, int D,
                           const float* __restrict__ h, const float* __restrict__ wbuf,
                           const float* __restrict__ ssum, float* __restrict__ outp) {
    long long wid = (long long)blockIdx.x * (blockDim.x >> 5) + (threadIdx.x >> 5);
    if (wid >= E) return;
    int lane = threadIdx.x & 31;
    int src, dst; edge_nodes(ei, E0, wid, src, dst);
    const float* hp = h + (size_t)src * F;
    float* op = outp + (size_t)dst * F;
    for (int f = lane * 4; f < F; f += 128) {
        int hh = f / D;
        float wt = wbuf[(size_t)wid * H + hh] / (ssum[(size_t)dst * H + hh] + 1e-16f);
        float4 v = *(const float4*)(hp + f);
        atomicAdd(op + f + 0, v.x * wt);
        atomicAdd(op + f + 1, v.y * wt);
        atomicAdd(op + f + 2, v.z * wt);
        atomicAdd(op + f + 3, v.w * wt);
    }
}

// ---------- bias + ELU in place ----------
__global__ void k_bias_elu(float* __restrict__ a, const float* __restrict__ b, int N, int F) {
    size_t t = (size_t)blockIdx.x * blockDim.x + threadIdx.x;
    if (t >= (size_t)N * F) return;
    float v = a[t] + b[t % F];
    a[t] = v > 0.f ? v : (__expf(v) - 1.f);
}

// ---------- bias + L2 normalize rows in place (wave per node, F=128) ----------
__global__ void k_normalize(float* __restrict__ o, const float* __restrict__ b, int N, int F) {
    long long n = (long long)blockIdx.x * (blockDim.x >> 5) + (threadIdx.x >> 5);
    if (n >= N) return;
    int lane = threadIdx.x & 31;
    float v[4]; float ss = 0.f;
#pragma unroll
    for (int j = 0; j < 4; ++j) {
        int f = lane * 4 + j;
        v[j] = o[(size_t)n * F + f] + b[f];
        ss += v[j] * v[j];
    }
#pragma unroll
    for (int m = 16; m >= 1; m >>= 1) ss += __shfl_xor(ss, m, 32);
    float inv = 1.f / fmaxf(__builtin_sqrtf(ss), 1e-12f);
#pragma unroll
    for (int j = 0; j < 4; ++j) o[(size_t)n * F + lane * 4 + j] = v[j] * inv;
}

// ---------------------------------------------------------------------------
extern "C" void kernel_launch(void* const* d_in, const int* in_sizes, int n_in,
                              void* d_out, int out_size, void* d_ws, size_t ws_size,
                              hipStream_t stream) {
    const float*     x    = (const float*)d_in[0];
    const long long* ei   = (const long long*)d_in[1];
    const float*     W1   = (const float*)d_in[2];
    const float*     aS1  = (const float*)d_in[3];
    const float*     aD1  = (const float*)d_in[4];
    const float*     b1   = (const float*)d_in[5];
    const float*     W2   = (const float*)d_in[6];
    const float*     aS2  = (const float*)d_in[7];
    const float*     aD2  = (const float*)d_in[8];
    const float*     b2   = (const float*)d_in[9];
    float* out = (float*)d_out;

    const long long N  = in_sizes[0] / IN_DIM;
    const long long E0 = in_sizes[1] / 2;
    const long long E  = E0 + N;

    // workspace carve-out (256B aligned)
    char* base = (char*)d_ws;
    size_t off = 0;
    auto carve = [&](size_t bytes) -> char* {
        off = (off + 255) & ~(size_t)255;
        char* p = base + off;
        off += bytes;
        return p;
    };
    float*          h1    = (float*)carve((size_t)N * F1 * 4);
    float*          agg1  = (float*)carve((size_t)N * F1 * 4);   // -> elu -> feat2
    float*          h2    = (float*)carve((size_t)N * OUT_DIM * 4);
    float*          as1   = (float*)carve((size_t)N * HEADS * 4);
    float*          ad1   = (float*)carve((size_t)N * HEADS * 4);
    unsigned*       mk1   = (unsigned*)carve((size_t)N * HEADS * 4);
    float*          s1    = (float*)carve((size_t)N * HEADS * 4);
    float*          wb1   = (float*)carve((size_t)E * HEADS * 4);
    float*          as2   = (float*)carve((size_t)N * 4);
    float*          ad2   = (float*)carve((size_t)N * 4);
    unsigned*       mk2   = (unsigned*)carve((size_t)N * 4);
    float*          s2    = (float*)carve((size_t)N * 4);
    float*          wb2   = (float*)carve((size_t)E * 4);
    unsigned short* W1T   = (unsigned short*)carve((size_t)F1 * IN_DIM * 2);
    unsigned short* W2T   = (unsigned short*)carve((size_t)OUT_DIM * F1 * 2);
    (void)ws_size; (void)n_in; (void)out_size;

    const unsigned ENC_NEG_INF = 0x007FFFFFu;  // enc(-inf)
    const int TB = 256;
    auto g1 = [&](long long n) { return dim3((unsigned)((n + TB - 1) / TB)); };

    // 0) weight transpose+convert
    k_transpose_bf16<<<g1((long long)IN_DIM * F1), TB, 0, stream>>>(W1, W1T, IN_DIM, F1);
    k_transpose_bf16<<<g1((long long)F1 * OUT_DIM), TB, 0, stream>>>(W2, W2T, F1, OUT_DIM);

    // 1) GEMM1: h1 = x @ W1   [N,1024]x[1024,256]
    {
        dim3 grid((unsigned)((N + 127) / 128), F1 / 128);
        k_gemm_bf16_wmma<<<grid, TB, 0, stream>>>(x, W1T, h1, (int)N, IN_DIM, F1);
    }
    // 2) attention halves layer 1
    k_alpha<<<g1(N * HEADS), TB, 0, stream>>>(h1, aS1, aD1, as1, ad1, (int)N, HEADS, HID);
    // 3) init scratch
    k_fill_u32<<<dim3(1024), TB, 0, stream>>>(mk1, ENC_NEG_INF, (size_t)N * HEADS);
    k_fill_f32<<<dim3(1024), TB, 0, stream>>>(s1, 0.f, (size_t)N * HEADS);
    k_fill_f32<<<dim3(4096), TB, 0, stream>>>(agg1, 0.f, (size_t)N * F1);
    // 4) edge softmax + aggregate, layer 1
    k_edge_max   <<<g1(E * HEADS), TB, 0, stream>>>(ei, E0, E, HEADS, as1, ad1, mk1);
    k_edge_expsum<<<g1(E * HEADS), TB, 0, stream>>>(ei, E0, E, HEADS, as1, ad1, mk1, wb1, s1);
    k_edge_agg   <<<g1(E * 32),    TB, 0, stream>>>(ei, E0, E, HEADS, F1, HID, h1, wb1, s1, agg1);
    // 5) bias + ELU -> feat2 (in place)
    k_bias_elu<<<g1(N * F1), TB, 0, stream>>>(agg1, b1, (int)N, F1);

    // 6) GEMM2: h2 = feat2 @ W2   [N,256]x[256,128]
    {
        dim3 grid((unsigned)((N + 127) / 128), OUT_DIM / 128);
        k_gemm_bf16_wmma<<<grid, TB, 0, stream>>>(agg1, W2T, h2, (int)N, F1, OUT_DIM);
    }
    // 7) attention halves layer 2 (H=1, D=128)
    k_alpha<<<g1(N), TB, 0, stream>>>(h2, aS2, aD2, as2, ad2, (int)N, 1, OUT_DIM);
    // 8) init scratch + output accumulator
    k_fill_u32<<<dim3(512), TB, 0, stream>>>(mk2, ENC_NEG_INF, (size_t)N);
    k_fill_f32<<<dim3(512), TB, 0, stream>>>(s2, 0.f, (size_t)N);
    k_fill_f32<<<dim3(4096), TB, 0, stream>>>(out, 0.f, (size_t)N * OUT_DIM);
    // 9) edge softmax + aggregate, layer 2 (into d_out)
    k_edge_max   <<<g1(E), TB, 0, stream>>>(ei, E0, E, 1, as2, ad2, mk2);
    k_edge_expsum<<<g1(E), TB, 0, stream>>>(ei, E0, E, 1, as2, ad2, mk2, wb2, s2);
    k_edge_agg   <<<g1(E * 32), TB, 0, stream>>>(ei, E0, E, 1, OUT_DIM, OUT_DIM, h2, wb2, s2, out);
    // 10) bias + L2 normalize rows in place
    k_normalize<<<g1(N * 32), TB, 0, stream>>>(out, b2, (int)N, OUT_DIM);
}
